// PretrainModel_11304353923870
// MI455X (gfx1250) — matrix operand order repaired and checked by reading, not compiled, not run on hardware
//
#include <hip/hip_runtime.h>

// ---------------------------------------------------------------------------
// GIN layer for MI455X (gfx1250, wave32, WMMA bf16 16x16x32, f32 accum).
//   1. h0f = x ; scatter-add x[src] into h0f[dst]   (f32 atomics, mem-bound)
//   2. h0b = bf16(h0f)                              (packed RNE convert)
//   3. h1  = relu(h0b @ W1 + b1)                    (WMMA, double-buffered LDS)
//   4. h2  = relu(h1  @ W2 + b2)
//   5. out[g] += h2 @ Wo + bo                       (fused global_add_pool)
// ---------------------------------------------------------------------------

#define NN 100000   // nodes
#define NE 1600000  // edges
#define NG 512      // graphs
#define DIN 128
#define DHID 256
#define DOUT 128

typedef __attribute__((ext_vector_type(16))) __bf16 v16bf;
typedef __attribute__((ext_vector_type(8)))  float  v8f;

struct U32x8 { uint4 lo, hi; };  // 32B container for one v16bf fragment

__device__ __forceinline__ unsigned short f2bf(float f) {
    unsigned int u = __float_as_uint(f);
    u += 0x7FFFu + ((u >> 16) & 1u);          // round-to-nearest-even
    return (unsigned short)(u >> 16);
}

// two f32 -> packed bf16x2 (all 32-bit integer math, no 16-bit subregs)
__device__ __forceinline__ unsigned int pack2bf(float lo, float hi) {
    unsigned int a = __float_as_uint(lo);
    unsigned int b = __float_as_uint(hi);
    a = (a + 0x7FFFu + ((a >> 16) & 1u)) >> 16;
    b = (b + 0x7FFFu + ((b >> 16) & 1u)) & 0xFFFF0000u;
    return (a & 0xFFFFu) | b;
}

// ---------------- small utility kernels ----------------
__global__ void zero_f32(float* __restrict__ p, int n) {
    int i = blockIdx.x * 256 + threadIdx.x;
    if (i < n) p[i] = 0.0f;
}

__global__ void copy_f32x4(const float4* __restrict__ s, float4* __restrict__ d, int n4) {
    int i = blockIdx.x * 256 + threadIdx.x;
    if (i < n4) d[i] = s[i];
}

// f32 -> bf16, 8 elements per thread (2x b128 load, 1x b128 store)
__global__ void cvt_f32_bf16(const float4* __restrict__ s, uint4* __restrict__ d, int n8) {
    int i = blockIdx.x * 256 + threadIdx.x;
    if (i >= n8) return;
    float4 f0 = s[2 * i + 0];
    float4 f1 = s[2 * i + 1];
    uint4 o;
    o.x = pack2bf(f0.x, f0.y);
    o.y = pack2bf(f0.z, f0.w);
    o.z = pack2bf(f1.x, f1.y);
    o.w = pack2bf(f1.z, f1.w);
    d[i] = o;
}

// scatter-add: h0[dst] += x[src], one (edge, 4-dim chunk) per thread
__global__ void edge_scatter(const float* __restrict__ x, const int* __restrict__ ei,
                             float* __restrict__ h0) {
    long gid = (long)blockIdx.x * 256 + threadIdx.x;   // NE*32 threads
    int e = (int)(gid >> 5);
    int c = (int)(gid & 31) << 2;
    if (e >= NE) return;
    int src = ei[e];
    int dst = ei[NE + e];
    const float4 v = *(const float4*)(x + (unsigned)src * DIN + c);
    float* p = h0 + (unsigned)dst * DIN + c;
    atomicAdd(p + 0, v.x);
    atomicAdd(p + 1, v.y);
    atomicAdd(p + 2, v.z);
    atomicAdd(p + 3, v.w);
}

// W[K][N] f32  ->  Wt[N][K] bf16 (transposed so B fragments are K-contiguous)
__global__ void wcvt(const float* __restrict__ W, unsigned short* __restrict__ Wt,
                     int K, int N) {
    int idx = blockIdx.x * 256 + threadIdx.x;
    if (idx >= K * N) return;
    int n = idx / K, k = idx - n * K;
    Wt[idx] = f2bf(W[(unsigned)k * N + n]);
}

// ---------------- WMMA GEMM: C = act(A @ Wt^T + b), optionally pooled -------
// A: [NN][K] bf16, Wt: [N][K] bf16, block tile 128(M) x 64(N), 8 waves,
// each wave 32x32 via 2x2 v_wmma_f32_16x16x32_bf16. Double-buffered LDS,
// K/N templated so the step loop fully unrolls.
// Global loads are UNCONDITIONAL: A row index is clamped to NN-1 (garbage
// rows only feed output rows >= NN which the epilogue discards), and N is a
// multiple of 64 so B never needs a guard. With base + zext(32-bit voffset)
// + constant, the backend emits global_load_b128 saddr form (LOADcnt only).
template <bool RELU, bool POOL, int K, int N>
__global__ __launch_bounds__(256)
void gin_gemm(const unsigned short* __restrict__ A, const unsigned short* __restrict__ Wt,
              const float* __restrict__ bias, unsigned short* __restrict__ Obf,
              float* __restrict__ Opool, const int* __restrict__ batch) {
    __shared__ __align__(64) unsigned short As[2][128 * 32];  // [m][k]
    __shared__ __align__(64) unsigned short Bs[2][64 * 32];   // [n][k]

    const int t    = threadIdx.x;
    const int m0   = blockIdx.x * 128;
    const int n0   = blockIdx.y * 64;
    const int w    = t >> 5;
    const int lane = t & 31;
    const int wm   = (w & 3) * 32;   // wave M offset in tile
    const int wn   = (w >> 2) * 32;  // wave N offset in tile
    const int lh   = lane >> 4;      // lane half (0/1)
    const int lm   = lane & 15;

    // per-thread staging coordinates (16B chunks)
    const int ra0 = t >> 2;             // A rows 0..63
    const int ra1 = 64 + (t >> 2);      // A rows 64..127
    const int ca  = (t & 3) * 8;        // k-chunk within 32
    const int rb  = t >> 2;             // B rows 0..63

    // 32-bit element base offsets, rows clamped in-bounds (no guards needed)
    const unsigned oa0 = (unsigned)min(m0 + ra0, NN - 1) * K + (unsigned)ca;
    const unsigned oa1 = (unsigned)min(m0 + ra1, NN - 1) * K + (unsigned)ca;
    const unsigned ob  = (unsigned)(n0 + rb) * K + (unsigned)ca;

    v8f acc[2][2];
#pragma unroll
    for (int i = 0; i < 2; ++i)
#pragma unroll
        for (int j = 0; j < 2; ++j)
#pragma unroll
            for (int r = 0; r < 8; ++r) acc[i][j][r] = 0.0f;

    uint4 va0, va1, vb;

    auto load_tile = [&](int kk) {           // kk constant per unrolled step
        va0 = *(const uint4*)(A  + oa0 + kk);
        va1 = *(const uint4*)(A  + oa1 + kk);
        vb  = *(const uint4*)(Wt + ob  + kk);
    };
    auto store_tile = [&](int buf) {
        *(uint4*)&As[buf][ra0 * 32 + ca] = va0;
        *(uint4*)&As[buf][ra1 * 32 + ca] = va1;
        *(uint4*)&Bs[buf][rb  * 32 + ca] = vb;
    };

    // prologue: tile 0 into buffer 0
    load_tile(0);
    store_tile(0);
    __syncthreads();

    constexpr int NSTEPS = K >> 5;
#pragma unroll
    for (int s = 0; s < NSTEPS; ++s) {
        const int buf = s & 1;                     // constant after unroll
        const bool hasnext = (s + 1) < NSTEPS;     // constant after unroll
        if (hasnext) load_tile((s + 1) << 5);      // overlap with WMMAs below

        // ---- fragments (ISA 7.12.2 layouts) + WMMAs from LDS[buf]
        v16bf afr[2], bfr[2];
#pragma unroll
        for (int i = 0; i < 2; ++i) {
            int row = wm + i * 16 + lm;          // A: lane holds row lm
            U32x8 u;
            u.lo = *(const uint4*)&As[buf][row * 32 + lh * 8];       // K lo
            u.hi = *(const uint4*)&As[buf][row * 32 + 16 + lh * 8];  // K hi
            afr[i] = __builtin_bit_cast(v16bf, u);
        }
#pragma unroll
        for (int j = 0; j < 2; ++j) {
            int row = wn + j * 16 + lm;          // B: lane holds column lm
            U32x8 u;                             // K lo/hi halves, same as A
            u.lo = *(const uint4*)&Bs[buf][row * 32 + lh * 8];
            u.hi = *(const uint4*)&Bs[buf][row * 32 + 16 + lh * 8];
            bfr[j] = __builtin_bit_cast(v16bf, u);
        }
#pragma unroll
        for (int i = 0; i < 2; ++i)
#pragma unroll
            for (int j = 0; j < 2; ++j)
                acc[i][j] = __builtin_amdgcn_wmma_f32_16x16x32_bf16(
                    false, afr[i], false, bfr[j], (short)0, acc[i][j],
                    false, false);

        if (hasnext) {
            store_tile(buf ^ 1);   // other waves may still read LDS[buf]: OK
            __syncthreads();       // one barrier per K-step
        }
    }

    // ---- epilogue: C/D layout n = lane%16, m = (lane/16)*8 + r
#pragma unroll
    for (int i = 0; i < 2; ++i) {
#pragma unroll
        for (int j = 0; j < 2; ++j) {
            int gn = n0 + wn + j * 16 + lm;
            float bv = bias[gn];
#pragma unroll
            for (int r = 0; r < 8; ++r) {
                int gm = m0 + wm + i * 16 + lh * 8 + r;
                if (gm < NN) {
                    float v = acc[i][j][r] + bv;
                    if constexpr (RELU) v = fmaxf(v, 0.0f);
                    if constexpr (POOL) {
                        int g = batch[gm];
                        atomicAdd(Opool + (unsigned)g * N + gn, v);
                    } else {
                        Obf[(unsigned)gm * N + gn] = f2bf(v);
                    }
                }
            }
        }
    }
}

// ---------------------------------------------------------------------------
extern "C" void kernel_launch(void* const* d_in, const int* in_sizes, int n_in,
                              void* d_out, int out_size, void* d_ws, size_t ws_size,
                              hipStream_t stream) {
    (void)in_sizes; (void)n_in; (void)out_size; (void)ws_size;

    const float* x     = (const float*)d_in[0];
    const int*   ei    = (const int*)  d_in[1];   // [2][NE] (JAX default int32)
    const int*   batch = (const int*)  d_in[2];
    const float* W1    = (const float*)d_in[3];
    const float* b1    = (const float*)d_in[4];
    const float* W2    = (const float*)d_in[5];
    const float* b2    = (const float*)d_in[6];
    const float* Wo    = (const float*)d_in[7];
    const float* bo    = (const float*)d_in[8];
    float* out = (float*)d_out;                   // [NG][DOUT]

    // workspace layout (~128.3 MB)
    char* ws = (char*)d_ws;
    float*          h0f = (float*)(ws + 0);                      // f32  [NN][128]
    unsigned short* h2  = (unsigned short*)(ws + 0);             // bf16 [NN][256], reuses h0f
    unsigned short* h0b = (unsigned short*)(ws + 51200000);      // bf16 [NN][128]
    unsigned short* h1  = (unsigned short*)(ws + 76800000);      // bf16 [NN][256]
    unsigned short* W1t = (unsigned short*)(ws + 128000000);     // [256][128]
    unsigned short* W2t = (unsigned short*)(ws + 128065536);     // [256][256]
    unsigned short* Wot = (unsigned short*)(ws + 128196608);     // [128][256]

    // 0) zero pooled output (accumulated via atomics every call)
    zero_f32<<<(NG * DOUT + 255) / 256, 256, 0, stream>>>(out, NG * DOUT);

    // 1) h0f = x ; then h0f[dst] += x[src] over all edges
    copy_f32x4<<<(NN * DIN / 4 + 255) / 256, 256, 0, stream>>>(
        (const float4*)x, (float4*)h0f, NN * DIN / 4);
    edge_scatter<<<(int)(((long)NE * 32 + 255) / 256), 256, 0, stream>>>(x, ei, h0f);

    // 2) h0b = bf16(h0f); weights -> bf16 [N][K]
    cvt_f32_bf16<<<(NN * DIN / 8 + 255) / 256, 256, 0, stream>>>(
        (const float4*)h0f, (uint4*)h0b, NN * DIN / 8);
    wcvt<<<(DIN * DHID + 255) / 256, 256, 0, stream>>>(W1, W1t, DIN, DHID);
    wcvt<<<(DHID * DHID + 255) / 256, 256, 0, stream>>>(W2, W2t, DHID, DHID);
    wcvt<<<(DHID * DOUT + 255) / 256, 256, 0, stream>>>(Wo, Wot, DHID, DOUT);

    const int mblk = (NN + 127) / 128;  // 782

    // 3) h1 = relu(h0b @ W1 + b1)
    gin_gemm<true, false, DIN, DHID><<<dim3(mblk, DHID / 64), 256, 0, stream>>>(
        h0b, W1t, b1, h1, nullptr, nullptr);

    // 4) h2 = relu(h1 @ W2 + b2)
    gin_gemm<true, false, DHID, DHID><<<dim3(mblk, DHID / 64), 256, 0, stream>>>(
        h1, W2t, b2, h2, nullptr, nullptr);

    // 5) out[g] += h2 @ Wo + bo (fused global_add_pool)
    gin_gemm<false, true, DHID, DOUT><<<dim3(mblk, DOUT / 64), 256, 0, stream>>>(
        h2, Wot, bo, nullptr, out, batch);
}
